// GNNComplete_82978768159420
// MI455X (gfx1250) — matrix operand order repaired and checked by reading, not compiled, not run on hardware
//
#include <hip/hip_runtime.h>

#define NUM_ENT 100000
#define NE      500000
#define DIM     200
#define DIMP    224        // 200 padded to x32
#define HALF    100
#define HALFP   112        // 100 padded to x16
#define NREL2   474
#define NRELP   480        // 474 padded to x16
#define DIMNP   208        // 200 padded to x16 (N side of 200-col weights)
#define NT      131072
#define CH      16384
#define NCH     (NT / CH)
#define CATP    896        // 875 padded to x32
#define CATK    875
#define H1N     2048
#define H2N     512

typedef __attribute__((ext_vector_type(16))) __bf16 bf16x16;
typedef __attribute__((ext_vector_type(8)))  __bf16 bf16x8;
typedef __attribute__((ext_vector_type(8)))  float  f32x8;

// ---------------------------------------------------------------------------
// WMMA core: A row-major bf16 (lda, K padded x32), B transposed bf16
// Bt[n][k] (ldb). One A fragment (2x16B loads) reused across NB column tiles
// (each one 32B load + one v_wmma_f32_16x16x32_bf16). No branches in loop.
// A lane layout: row=lane&15, k = hi*16 + {0..15}
// B lane layout: col=lane&15, k = hi*16 + {0..15}
// C/D: col=lane&15, rows = hi*8 + v
// ---------------------------------------------------------------------------
template <int NB>
__device__ __forceinline__ void wmma_core_p(const __bf16* __restrict__ ap,
                                            const __bf16* __restrict__ bp,
                                            long ldb, int K, f32x8* acc)
{
    for (int k0 = 0; k0 < K; k0 += 32) {
        bf16x8 a0 = *(const bf16x8*)(ap + k0);
        bf16x8 a1 = *(const bf16x8*)(ap + k0 + 8);
        bf16x16 a = __builtin_shufflevector(a0, a1, 0, 1, 2, 3, 4, 5, 6, 7,
                                            8, 9, 10, 11, 12, 13, 14, 15);
#pragma unroll
        for (int j = 0; j < NB; ++j) {
            bf16x16 b = *(const bf16x16*)(bp + (long)(j * 16) * ldb + k0);
            acc[j] = __builtin_amdgcn_wmma_f32_16x16x32_bf16(false, a, false, b,
                                                             (short)0, acc[j],
                                                             false, false);
        }
    }
}

template <int NB>
__device__ __forceinline__ void wmma_core(const __bf16* __restrict__ A, long lda,
                                          const __bf16* __restrict__ Bt, long ldb,
                                          int row0, int col0, int K, f32x8* acc)
{
    const int lane = threadIdx.x & 31;
    const int hi = lane >> 4, lo = lane & 15;
    const __bf16* ap = A + (long)(row0 + lo) * lda + (hi << 4);
    const __bf16* bp = Bt + (long)(col0 + lo) * ldb + (hi << 4);
    wmma_core_p<NB>(ap, bp, ldb, K, acc);
}

// ---------------------------------------------------------------------------
__global__ void k_zero(float* p, long n)
{
    long i = (long)blockIdx.x * blockDim.x + threadIdx.x;
    if (i < n) p[i] = 0.0f;
}

// f32 (M x K) row-major -> bf16 (Mp x Kp) row-major, zero padded
__global__ void k_cvt_a(const float* __restrict__ src, __bf16* __restrict__ dst,
                        int M, int K, int Kp, long total)
{
    long idx = (long)blockIdx.x * blockDim.x + threadIdx.x;
    if (idx >= total) return;
    int r = (int)(idx / Kp), k = (int)(idx % Kp);
    dst[idx] = (r < M && k < K) ? (__bf16)src[(long)r * K + k] : (__bf16)0.0f;
}

// f32 weight (K x N) row-major -> bf16 transposed (Np x Kp), zero padded
__global__ void k_cvt_w(const float* __restrict__ src, __bf16* __restrict__ dst,
                        int K, int N, int Kp, long total)
{
    long idx = (long)blockIdx.x * blockDim.x + threadIdx.x;
    if (idx >= total) return;
    int n = (int)(idx / Kp), k = (int)(idx % Kp);
    dst[idx] = (k < K && (long)n < (total / Kp) && n < N)
                   ? (__bf16)src[(long)k * N + n] : (__bf16)0.0f;
}

// S[dst] += ent_emb[src] - rel_emb[type]; deg[dst] += 1
__global__ void k_edge_scatter(const int* __restrict__ ei, const int* __restrict__ et,
                               const float* __restrict__ ent, const float* __restrict__ rel,
                               float* __restrict__ S, float* __restrict__ deg)
{
    long idx = (long)blockIdx.x * blockDim.x + threadIdx.x;
    if (idx >= (long)NE * 50) return;
    int e = (int)(idx / 50), g = (int)(idx % 50);
    int src = ei[e], dst = ei[NE + e], t = et[e];
    const float* es = ent + (long)src * DIM + g * 4;
    const float* rs = rel + (long)t   * DIM + g * 4;
    float*       sp = S   + (long)dst * DIM + g * 4;
#pragma unroll
    for (int d = 0; d < 4; ++d) atomicAdd(sp + d, es[d] - rs[d]);
    if (g == 0) atomicAdd(deg + dst, 1.0f);
}

__global__ void k_norm(float* deg_norm)
{
    int i = blockIdx.x * blockDim.x + threadIdx.x;
    if (i < NUM_ENT) deg_norm[i] = rsqrtf(fmaxf(deg_norm[i], 1.0f));
}

// aentb = bf16(tanh(norm[r]*(S@W) + E@W_loop))  (M=100000, K=224, N=208 tiles)
__global__ __launch_bounds__(128)
void k_all_ent(const __bf16* __restrict__ Sb, const __bf16* __restrict__ Wt,
               const __bf16* __restrict__ Eb, const __bf16* __restrict__ Wlt,
               const float* __restrict__ nrm, __bf16* __restrict__ aentb)
{
    const int Ntl = DIMNP / 16;                       // 13
    const int T   = (NUM_ENT / 16) * Ntl;
    int t = (blockIdx.x * blockDim.x + threadIdx.x) >> 5;
    if (t >= T) t = T - 1;
    const int row0 = (t / Ntl) * 16, col0 = (t % Ntl) * 16;
    f32x8 a1[1] = {{}}, a2[1] = {{}};
    wmma_core<1>(Sb, DIMP, Wt,  DIMP, row0, col0, DIMP, a1);
    wmma_core<1>(Eb, DIMP, Wlt, DIMP, row0, col0, DIMP, a2);
    const int lane = threadIdx.x & 31, hi = lane >> 4, lo = lane & 15;
    const int c = col0 + lo;
#pragma unroll
    for (int v = 0; v < 8; ++v) {
        int r = row0 + (hi << 3) + v;
        aentb[(long)r * DIMP + c] = (__bf16)tanhf(nrm[r] * a1[0][v] + a2[0][v]);
    }
}

// arelb[n][k] = +-bf16(rel_emb@W_rel)[n][k]  (sign-folded head/tail, Bt layout)
__global__ __launch_bounds__(128)
void k_all_rel(const __bf16* __restrict__ relb, const __bf16* __restrict__ Wrt,
               __bf16* __restrict__ arelb)
{
    const int Ntl = DIMNP / 16;                       // 13
    const int T   = (NRELP / 16) * Ntl;               // 30*13
    int t = (blockIdx.x * blockDim.x + threadIdx.x) >> 5;
    if (t >= T) t = T - 1;
    const int row0 = (t / Ntl) * 16, col0 = (t % Ntl) * 16;
    f32x8 acc[1] = {{}};
    wmma_core<1>(relb, DIMP, Wrt, DIMP, row0, col0, DIMP, acc);
    const int lane = threadIdx.x & 31, hi = lane >> 4, lo = lane & 15;
    const int c = col0 + lo;
    const float sgn = (c < HALF) ? 1.0f : -1.0f;
#pragma unroll
    for (int v = 0; v < 8; ++v) {
        int r = row0 + (hi << 3) + v;
        arelb[(long)r * DIMP + c] = (__bf16)(sgn * acc[0][v]);
    }
}

// eeb[i][0:100]=h_e, [100:200]=t_e : gathered all_ent @ trans_w + trans_b
__global__ __launch_bounds__(128)
void k_ee(const __bf16* __restrict__ aentb, const int* __restrict__ toTest,
          const __bf16* __restrict__ twt, const float* __restrict__ tb,
          __bf16* __restrict__ eeb, int base2)
{
    const int Ntl = HALFP / 16;                       // 7
    const int T   = (2 * CH / 16) * Ntl;
    int t = (blockIdx.x * blockDim.x + threadIdx.x) >> 5;
    if (t >= T) t = T - 1;
    const int row0 = (t / Ntl) * 16, col0 = (t % Ntl) * 16;
    const int lane = threadIdx.x & 31, hi = lane >> 4, lo = lane & 15;
    const long entr = (long)toTest[base2 + row0 + lo];   // fixed per lane
    const __bf16* ap = aentb + entr * DIMP + (hi << 4);
    const __bf16* bp = twt + (long)(col0 + lo) * DIMP + (hi << 4);
    f32x8 acc[1] = {{}};
    wmma_core_p<1>(ap, bp, DIMP, DIMP, acc);
    const int c = col0 + lo;
#pragma unroll
    for (int v = 0; v < 8; ++v) {
        int r = row0 + (hi << 3) + v;
        if (c < HALF)
            eeb[(long)(r >> 1) * DIMP + (r & 1) * HALF + c] = (__bf16)(acc[0][v] + tb[c]);
    }
}

// atta[i] = dot(h_e@q_w+q_b, t_e@k_w+k_b) / sqrt(200)
__global__ void k_atta(const __bf16* __restrict__ eeb,
                       const float* __restrict__ qw, const float* __restrict__ qb,
                       const float* __restrict__ kw, const float* __restrict__ kb,
                       float* __restrict__ atta)
{
    const int i = blockIdx.x;
    const int n = threadIdx.x;                        // 0..127
    const __bf16* h = eeb + (long)i * DIMP;
    const __bf16* t = h + HALF;
    float q = qb[n], kk = kb[n];
    for (int k = 0; k < HALF; ++k) {
        q  += (float)h[k] * qw[k * 128 + n];
        kk += (float)t[k] * kw[k * 128 + n];
    }
    float p = q * kk;
    for (int off = 16; off; off >>= 1) p += __shfl_xor(p, off);
    __shared__ float sm[4];
    if ((n & 31) == 0) sm[n >> 5] = p;
    __syncthreads();
    if (n == 0) atta[i] = (sm[0] + sm[1] + sm[2] + sm[3]) * 0.070710678f;
}

// relSim = [h_e|t_e] @ arelb^T  (sign pre-folded) -> bf16 cat[:,401:875]
__global__ __launch_bounds__(128)
void k_relsim(const __bf16* __restrict__ eeb, const __bf16* __restrict__ arelb,
              __bf16* __restrict__ cat)
{
    const int NG = NRELP / 32;                        // 15 groups of 2 tiles
    const int T  = (CH / 16) * NG;
    int t = (blockIdx.x * blockDim.x + threadIdx.x) >> 5;
    if (t >= T) t = T - 1;
    const int row0 = (t / NG) * 16, col0 = (t % NG) * 32;
    f32x8 acc[2] = {{}, {}};
    wmma_core<2>(eeb, DIMP, arelb, DIMP, row0, col0, DIMP, acc);
    const int lane = threadIdx.x & 31, hi = lane >> 4, lo = lane & 15;
#pragma unroll
    for (int j = 0; j < 2; ++j) {
        const int c = col0 + j * 16 + lo;
#pragma unroll
        for (int v = 0; v < 8; ++v) {
            int r = row0 + (hi << 3) + v;
            if (c < NREL2) cat[(long)r * CATP + 401 + c] = (__bf16)acc[j][v];
        }
    }
}

// cat[:,0:200]=te0, [:,200:400]=te1, [:,400]=atta  (bf16 copy from aentb)
__global__ void k_catfill(const __bf16* __restrict__ aentb, const int* __restrict__ toTest,
                          const float* __restrict__ atta, __bf16* __restrict__ cat, int base)
{
    long idx = (long)blockIdx.x * blockDim.x + threadIdx.x;
    if (idx >= (long)CH * 401) return;
    int r = (int)(idx / 401), c = (int)(idx % 401);
    __bf16* row = cat + (long)r * CATP;
    if (c < 400) {
        int which = c / DIM;
        int d = c - which * DIM;
        long entr = (long)toTest[2 * (base + r) + which];
        row[c] = aentb[entr * DIMP + d];
    } else {
        row[400] = (__bf16)atta[r];
    }
}

// h1 = lrelu(cat @ w1 + b1, 1e-4) bf16  (M=CH, K=896, N=2048; 16x64/wave)
__global__ __launch_bounds__(128)
void k_h1(const __bf16* __restrict__ cat, const __bf16* __restrict__ w1t,
          const float* __restrict__ b1, __bf16* __restrict__ h1)
{
    const int NG = H1N / 64;                          // 32
    const int T  = (CH / 16) * NG;
    int t = (blockIdx.x * blockDim.x + threadIdx.x) >> 5;
    if (t >= T) t = T - 1;
    const int row0 = (t / NG) * 16, col0 = (t % NG) * 64;
    f32x8 acc[4] = {{}, {}, {}, {}};
    wmma_core<4>(cat, CATP, w1t, CATP, row0, col0, CATP, acc);
    const int lane = threadIdx.x & 31, hi = lane >> 4, lo = lane & 15;
#pragma unroll
    for (int j = 0; j < 4; ++j) {
        const int c = col0 + j * 16 + lo;
        const float bias = b1[c];
#pragma unroll
        for (int v = 0; v < 8; ++v) {
            int r = row0 + (hi << 3) + v;
            float x = acc[j][v] + bias;
            h1[(long)r * H1N + c] = (__bf16)(x >= 0.0f ? x : 1e-4f * x);
        }
    }
}

// h2 = lrelu(h1 @ w2 + b2, 1e-3) f32    (M=CH, K=2048, N=512; 16x64/wave)
__global__ __launch_bounds__(128)
void k_h2(const __bf16* __restrict__ h1, const __bf16* __restrict__ w2t,
          const float* __restrict__ b2, float* __restrict__ h2)
{
    const int NG = H2N / 64;                          // 8
    const int T  = (CH / 16) * NG;
    int t = (blockIdx.x * blockDim.x + threadIdx.x) >> 5;
    if (t >= T) t = T - 1;
    const int row0 = (t / NG) * 16, col0 = (t % NG) * 64;
    f32x8 acc[4] = {{}, {}, {}, {}};
    wmma_core<4>(h1, H1N, w2t, H1N, row0, col0, H1N, acc);
    const int lane = threadIdx.x & 31, hi = lane >> 4, lo = lane & 15;
#pragma unroll
    for (int j = 0; j < 4; ++j) {
        const int c = col0 + j * 16 + lo;
        const float bias = b2[c];
#pragma unroll
        for (int v = 0; v < 8; ++v) {
            int r = row0 + (hi << 3) + v;
            float x = acc[j][v] + bias;
            h2[(long)r * H2N + c] = (x >= 0.0f ? x : 1e-3f * x);
        }
    }
}

// out = sigmoid(h2 @ w3 + b3)
__global__ void k_final(const float* __restrict__ h2, const float* __restrict__ w3,
                        const float* __restrict__ b3, float* __restrict__ out, int base)
{
    const int r = blockIdx.x;
    const int n = threadIdx.x;                        // 0..255
    const float* hr = h2 + (long)r * H2N;
    float s = hr[n] * w3[n] + hr[n + 256] * w3[n + 256];
    for (int off = 16; off; off >>= 1) s += __shfl_xor(s, off);
    __shared__ float sm[8];
    if ((n & 31) == 0) sm[n >> 5] = s;
    __syncthreads();
    if (n == 0) {
        float tot = b3[0];
#pragma unroll
        for (int i = 0; i < 8; ++i) tot += sm[i];
        out[base + r] = 1.0f / (1.0f + expf(-tot));
    }
}

// ---------------------------------------------------------------------------
static inline int gemm_blocks(int tiles) { return (tiles + 3) / 4; }   // 4 waves/block
static inline int nblk(long n) { return (int)((n + 255) / 256); }

extern "C" void kernel_launch(void* const* d_in, const int* in_sizes, int n_in,
                              void* d_out, int out_size, void* d_ws, size_t ws_size,
                              hipStream_t stream)
{
    const int*   ei  = (const int*)d_in[0];
    const int*   et  = (const int*)d_in[1];
    const int*   tT  = (const int*)d_in[2];
    const float* ent = (const float*)d_in[3];
    const float* rel = (const float*)d_in[4];
    const float* W   = (const float*)d_in[5];
    const float* Wl  = (const float*)d_in[6];
    const float* Wr  = (const float*)d_in[7];
    const float* tw  = (const float*)d_in[8];
    const float* tb  = (const float*)d_in[9];
    const float* qw  = (const float*)d_in[10];
    const float* qb  = (const float*)d_in[11];
    const float* kw  = (const float*)d_in[12];
    const float* kb  = (const float*)d_in[13];
    const float* w1  = (const float*)d_in[14];
    const float* b1  = (const float*)d_in[15];
    const float* w2  = (const float*)d_in[16];
    const float* b2  = (const float*)d_in[17];
    const float* w3  = (const float*)d_in[18];
    const float* b3  = (const float*)d_in[19];
    float* out = (float*)d_out;

    char*  ws = (char*)d_ws;
    size_t o  = 0;
    auto alloc = [&](size_t b) { size_t r = o; o += (b + 255) & ~(size_t)255; return r; };
    float*  nrm   = (float*)(ws + alloc((size_t)NUM_ENT * 4));
    float*  S     = (float*)(ws + alloc((size_t)NUM_ENT * DIM * 4));
    __bf16* Sb    = (__bf16*)(ws + alloc((size_t)NUM_ENT * DIMP * 2));
    __bf16* Eb    = (__bf16*)(ws + alloc((size_t)NUM_ENT * DIMP * 2));
    __bf16* aentb = (__bf16*)(ws + alloc((size_t)NUM_ENT * DIMP * 2));
    __bf16* relb  = (__bf16*)(ws + alloc((size_t)NRELP * DIMP * 2));
    __bf16* arelb = (__bf16*)(ws + alloc((size_t)NRELP * DIMP * 2));
    __bf16* Wt    = (__bf16*)(ws + alloc((size_t)DIMNP * DIMP * 2));
    __bf16* Wlt   = (__bf16*)(ws + alloc((size_t)DIMNP * DIMP * 2));
    __bf16* Wrt   = (__bf16*)(ws + alloc((size_t)DIMNP * DIMP * 2));
    __bf16* twt   = (__bf16*)(ws + alloc((size_t)HALFP * DIMP * 2));
    __bf16* w1t   = (__bf16*)(ws + alloc((size_t)H1N * CATP * 2));
    __bf16* w2t   = (__bf16*)(ws + alloc((size_t)H2N * H1N * 2));
    __bf16* eeb   = (__bf16*)(ws + alloc((size_t)CH * DIMP * 2));
    float*  atta  = (float*)(ws + alloc((size_t)CH * 4));
    __bf16* cat   = (__bf16*)(ws + alloc((size_t)CH * CATP * 2));
    __bf16* h1    = (__bf16*)(ws + alloc((size_t)CH * H1N * 2));
    float*  h2    = (float*)(ws + alloc((size_t)CH * H2N * 4));

    // ---- Phase 0: zero-init accumulators and padded buffers (f32-sized fills)
    k_zero<<<nblk(NUM_ENT), 256, 0, stream>>>(nrm, NUM_ENT);
    k_zero<<<nblk((long)NUM_ENT * DIM), 256, 0, stream>>>(S, (long)NUM_ENT * DIM);
    k_zero<<<nblk((long)NUM_ENT * DIMP / 2), 256, 0, stream>>>((float*)aentb, (long)NUM_ENT * DIMP / 2);
    k_zero<<<nblk((long)NRELP * DIMP / 2), 256, 0, stream>>>((float*)arelb, (long)NRELP * DIMP / 2);
    k_zero<<<nblk((long)CH * DIMP / 2), 256, 0, stream>>>((float*)eeb, (long)CH * DIMP / 2);
    k_zero<<<nblk((long)CH * CATP / 2), 256, 0, stream>>>((float*)cat, (long)CH * CATP / 2);

    // ---- Phase 0b: weight / embedding conversion to padded bf16 (Bt for weights)
    {
        long n;
        n = (long)DIMNP * DIMP;
        k_cvt_w<<<nblk(n), 256, 0, stream>>>(W,  Wt,  DIM, DIM, DIMP, n);
        k_cvt_w<<<nblk(n), 256, 0, stream>>>(Wl, Wlt, DIM, DIM, DIMP, n);
        k_cvt_w<<<nblk(n), 256, 0, stream>>>(Wr, Wrt, DIM, DIM, DIMP, n);
        n = (long)HALFP * DIMP;
        k_cvt_w<<<nblk(n), 256, 0, stream>>>(tw, twt, DIM, HALF, DIMP, n);
        n = (long)H1N * CATP;
        k_cvt_w<<<nblk(n), 256, 0, stream>>>(w1, w1t, CATK, H1N, CATP, n);
        n = (long)H2N * H1N;
        k_cvt_w<<<nblk(n), 256, 0, stream>>>(w2, w2t, H1N, H2N, H1N, n);
        n = (long)NUM_ENT * DIMP;
        k_cvt_a<<<nblk(n), 256, 0, stream>>>(ent, Eb, NUM_ENT, DIM, DIMP, n);
        n = (long)NRELP * DIMP;
        k_cvt_a<<<nblk(n), 256, 0, stream>>>(rel, relb, NREL2, DIM, DIMP, n);
    }

    // ---- Phase 1: graph aggregation (W folded out of the segment sum)
    {
        long nw = (long)NE * 50;
        k_edge_scatter<<<nblk(nw), 256, 0, stream>>>(ei, et, ent, rel, S, nrm);
        k_norm<<<nblk(NUM_ENT), 256, 0, stream>>>(nrm);
        long n = (long)NUM_ENT * DIMP;
        k_cvt_a<<<nblk(n), 256, 0, stream>>>(S, Sb, NUM_ENT, DIM, DIMP, n);
    }

    // ---- Phase 2: node / relation transforms (WMMA)
    {
        int T1 = (NUM_ENT / 16) * (DIMNP / 16);
        k_all_ent<<<gemm_blocks(T1), 128, 0, stream>>>(Sb, Wt, Eb, Wlt, nrm, aentb);
        int T2 = (NRELP / 16) * (DIMNP / 16);
        k_all_rel<<<gemm_blocks(T2), 128, 0, stream>>>(relb, Wrt, arelb);
    }

    // ---- Phase 3: chunked scoring pipeline (WMMA MLP)
    for (int c = 0; c < NCH; ++c) {
        int Tee = (2 * CH / 16) * (HALFP / 16);
        k_ee<<<gemm_blocks(Tee), 128, 0, stream>>>(aentb, tT, twt, tb, eeb, c * 2 * CH);
        k_atta<<<CH, 128, 0, stream>>>(eeb, qw, qb, kw, kb, atta);
        int Trs = (CH / 16) * (NRELP / 32);
        k_relsim<<<gemm_blocks(Trs), 128, 0, stream>>>(eeb, arelb, cat);
        long nc = (long)CH * 401;
        k_catfill<<<nblk(nc), 256, 0, stream>>>(aentb, tT, atta, cat, c * CH);
        int Th1 = (CH / 16) * (H1N / 64);
        k_h1<<<gemm_blocks(Th1), 128, 0, stream>>>(cat, w1t, b1, h1);
        int Th2 = (CH / 16) * (H2N / 64);
        k_h2<<<gemm_blocks(Th2), 128, 0, stream>>>(h1, w2t, b2, h2);
        k_final<<<CH, 256, 0, stream>>>(h2, w3, b3, out, c * CH);
    }
}
